// VisionTransformer_17910013624387
// MI455X (gfx1250) — matrix operand order repaired and checked by reading, not compiled, not run on hardware
//
#include <hip/hip_runtime.h>
#include <cmath>

// ---------------------------------------------------------------------------
// ViT-Base fwd on MI455X (gfx1250): all GEMMs via V_WMMA_F32_16X16X32_BF16,
// fp32 global data -> packed bf16 at LDS staging (v_perm pack), fp32 accum,
// software-pipelined global->LDS staging.
// ---------------------------------------------------------------------------

typedef __bf16 bf16;
typedef __attribute__((ext_vector_type(8)))  bf16  v8bf;
typedef __attribute__((ext_vector_type(16))) bf16  v16bf;
typedef __attribute__((ext_vector_type(8)))  float v8f;

#define BMT 128   // macro tile M
#define BNT 128   // macro tile N
#define KT  32    // k step (= WMMA K)
#define KTP 48    // padded LDS row (96B: keeps 4B/16B/32B alignment)

// pack two floats to packed bf16 (round-to-nearest) with a single v_perm_b32
__device__ __forceinline__ unsigned pk2(float lo, float hi) {
  unsigned ul = __builtin_bit_cast(unsigned, lo) + 0x8000u;
  unsigned uh = __builtin_bit_cast(unsigned, hi) + 0x8000u;
  return __builtin_amdgcn_perm(uh, ul, 0x07060302u);  // {uh[31:16], ul[31:16]}
}

struct GemmArgs {
  const float* A;      // A[m][k] (or images for MODE 3)
  const float* B;      // Bt[n][k] = B[k*sBk + n*sBn]
  const float* bias;   // per-n, nullable
  const float* resid;  // per (m,n), nullable (MODE 2)
  float*       C;
  long sAm, sBk, sBn, sRm, sCm;
  long zAb, zAh, zBb, zBh, zCb, zCh;   // batched (z = zb*Hdiv + zh) base strides
  int  M, N, K, Hdiv;
  float scale;
};

union FragA { v16bf v; v8bf h[2]; };

// MODE: 0 plain (+bias,*scale)   1 +bias,gelu   2 +bias,+resid   3 patch-embed
template <int MODE>
__launch_bounds__(256)
__global__ void gemm_wmma(GemmArgs g) {
  __shared__ alignas(32) bf16 As[BMT][KTP];
  __shared__ alignas(32) bf16 Bs[BNT][KTP];

  const int tid  = threadIdx.x;
  const int lane = tid & 31;
  const int wid  = tid >> 5;
  const int wm   = wid & 3;          // 4 waves along M (32 rows each)
  const int wn   = wid >> 2;         // 2 waves along N (64 cols each)
  const int half = lane >> 4;
  const int l15  = lane & 15;

  const int m0 = blockIdx.x * BMT;
  const int n0 = blockIdx.y * BNT;
  const int z  = blockIdx.z;
  const int zb = z / g.Hdiv;
  const int zh = z % g.Hdiv;

  const float* __restrict__ Ap = g.A + (long)zb * g.zAb + (long)zh * g.zAh;
  const float* __restrict__ Bp = g.B + (long)zb * g.zBb + (long)zh * g.zBh;
  float* __restrict__ Cp       = g.C + (long)zb * g.zCb + (long)zh * g.zCh;

  const v8f zf = {0.f,0.f,0.f,0.f,0.f,0.f,0.f,0.f};
  v8f acc[2][4];
  for (int i = 0; i < 2; i++)
    for (int j = 0; j < 4; j++) acc[i][j] = zf;

  // A staging: 2 threads/row, 16 consecutive k each
  const int la_row = tid >> 1;
  const int la_k0  = (tid & 1) * 16;
  // B staging: each thread stages a 2(k) x 8(n) sub-block (k-pairs pack to b32)
  const int bk  = (tid >> 4) * 2;        // 0,2,...,30
  const int bn0 = (tid & 15) * 8;        // 0..120

  float ta[16];          // staged A values
  float tb0[8], tb1[8];  // staged B values for k=bk, bk+1

  auto loadA = [&](int k0) {
    const int m  = m0 + la_row;
    const int kb = k0 + la_k0;
    if (MODE == 3) {
      // m -> (b,p), k -> (c, r, col): 16 consecutive k == 16 contiguous pixels
      const int b  = m / 196, p = m % 196;
      const int py = p / 14,  px = p % 14;
      const int c  = kb >> 8;
      const int rr = (kb >> 4) & 15;
      const float* src = Ap + (((long)(b * 3 + c) * 224 + py * 16 + rr) * 224 + px * 16);
      #pragma unroll
      for (int i = 0; i < 16; i++) ta[i] = src[i];
    } else {
      const float* src = Ap + (long)m * g.sAm + kb;
      if (m < g.M && kb + 16 <= g.K) {         // interior fast path: no selects
        #pragma unroll
        for (int i = 0; i < 16; i++) ta[i] = src[i];
      } else {
        #pragma unroll
        for (int i = 0; i < 16; i++)
          ta[i] = (m < g.M && (kb + i) < g.K) ? src[i] : 0.f;
      }
    }
  };
  auto loadB = [&](int k0) {
    const int k0a = k0 + bk, k0b = k0a + 1;
    const float* p0 = Bp + (long)k0a * g.sBk + (long)(n0 + bn0) * g.sBn;
    const float* p1 = p0 + g.sBk;
    if (k0b < g.K && n0 + bn0 + 8 <= g.N) {    // interior fast path
      #pragma unroll
      for (int i = 0; i < 8; i++) {
        tb0[i] = p0[(long)i * g.sBn];
        tb1[i] = p1[(long)i * g.sBn];
      }
    } else {
      #pragma unroll
      for (int i = 0; i < 8; i++) {
        const int n = n0 + bn0 + i;
        const bool nok = n < g.N;
        tb0[i] = (k0a < g.K && nok) ? p0[(long)i * g.sBn] : 0.f;
        tb1[i] = (k0b < g.K && nok) ? p1[(long)i * g.sBn] : 0.f;
      }
    }
  };

  loadA(0); loadB(0);

  int k0 = 0;
  while (true) {
    { // ---- pack bf16 pairs, vectorized LDS stores
      unsigned p[8];
      #pragma unroll
      for (int i = 0; i < 8; i++) p[i] = pk2(ta[2 * i], ta[2 * i + 1]);
      uint4* d = (uint4*)&As[la_row][la_k0];
      d[0] = make_uint4(p[0], p[1], p[2], p[3]);
      d[1] = make_uint4(p[4], p[5], p[6], p[7]);
      #pragma unroll
      for (int i = 0; i < 8; i++)
        *(unsigned*)&Bs[bn0 + i][bk] = pk2(tb0[i], tb1[i]);
    }
    __syncthreads();

    const int k1 = k0 + KT;
    if (k1 < g.K) { loadA(k1); loadB(k1); }   // in flight during the WMMAs

    #pragma unroll
    for (int mi = 0; mi < 2; mi++) {
      // A frag: lane m=l15; elems 0..15 -> k = 16h + e (contiguous)
      FragA a;
      const bf16* ar = &As[wm * 32 + mi * 16 + l15][0];
      a.h[0] = *(const v8bf*)(ar + 16 * half);
      a.h[1] = *(const v8bf*)(ar + 16 * half + 8);
      #pragma unroll
      for (int ni = 0; ni < 4; ni++) {
        // B frag: lane n=l15; elems 0..15 -> k = 16h + e (contiguous Bt row)
        const bf16* br = &Bs[wn * 64 + ni * 16 + l15][16 * half];
        v16bf bvec = *(const v16bf*)br;
        acc[mi][ni] = __builtin_amdgcn_wmma_f32_16x16x32_bf16(
            false, a.v, false, bvec, (short)0, acc[mi][ni], false, false);
      }
    }
    __syncthreads();
    k0 = k1;
    if (k0 >= g.K) break;
  }

  // ---- epilogue: D layout lane n=l15, vgpr r -> m = r + 8*half
  #pragma unroll
  for (int mi = 0; mi < 2; mi++) {
    #pragma unroll
    for (int r = 0; r < 8; r++) {
      const int m = m0 + wm * 32 + mi * 16 + r + 8 * half;
      if (m >= g.M) continue;
      #pragma unroll
      for (int ni = 0; ni < 4; ni++) {
        const int n = n0 + wn * 64 + ni * 16 + l15;
        if (n >= g.N) continue;
        float v = acc[mi][ni][r] * g.scale;
        if (g.bias) v += g.bias[n];
        if (MODE == 1) {                        // tanh-gelu (jax default)
          const float kAl = 0.7978845608028654f;
          v = 0.5f * v * (1.f + tanhf(kAl * (v + 0.044715f * v * v * v)));
        } else if (MODE == 2) {
          v += g.resid[(long)m * g.sRm + n];
        }
        if (MODE == 3) {                        // patch-embed: +pos, scatter x[b,1+p,:]
          const int b = m / 196, p = m % 196, s = p + 1;
          const float ang = (float)s * __powf(10000.f, -(float)n * (1.f / 768.f));
          v += (n & 1) ? __cosf(ang) : __sinf(ang);
          Cp[((long)(b * 197 + s)) * 768 + n] = v;
        } else {
          Cp[(long)m * g.sCm + n] = v;
        }
      }
    }
  }
}

// ---------------------------------------------------------------------------
__launch_bounds__(256)
__global__ void ln_kernel(const float* __restrict__ x, const float* __restrict__ gam,
                          const float* __restrict__ bet, float* __restrict__ h) {
  const long row = blockIdx.x;
  const int  tid = threadIdx.x;
  __shared__ float red[256];
  const float* xr = x + row * 768;

  float s = 0.f;
  for (int i = tid; i < 768; i += 256) s += xr[i];
  red[tid] = s; __syncthreads();
  for (int o = 128; o > 0; o >>= 1) { if (tid < o) red[tid] += red[tid + o]; __syncthreads(); }
  const float mean = red[0] * (1.f / 768.f);
  __syncthreads();

  float vs = 0.f;
  for (int i = tid; i < 768; i += 256) { float d = xr[i] - mean; vs += d * d; }
  red[tid] = vs; __syncthreads();
  for (int o = 128; o > 0; o >>= 1) { if (tid < o) red[tid] += red[tid + o]; __syncthreads(); }
  const float rstd = rsqrtf(red[0] * (1.f / 768.f) + 1e-5f);

  for (int i = tid; i < 768; i += 256)
    h[row * 768 + i] = (xr[i] - mean) * rstd * gam[i] + bet[i];
}

__launch_bounds__(256)
__global__ void softmax_rows(float* __restrict__ S, int ncols) {
  const long row = blockIdx.x;
  const int  tid = threadIdx.x;
  float* r = S + row * (long)ncols;
  __shared__ float red[256];

  float mx = -1e30f;
  for (int i = tid; i < ncols; i += 256) mx = fmaxf(mx, r[i]);
  red[tid] = mx; __syncthreads();
  for (int o = 128; o > 0; o >>= 1) { if (tid < o) red[tid] = fmaxf(red[tid], red[tid + o]); __syncthreads(); }
  const float m = red[0]; __syncthreads();

  float sum = 0.f;
  for (int i = tid; i < ncols; i += 256) { float e = __expf(r[i] - m); r[i] = e; sum += e; }
  red[tid] = sum; __syncthreads();
  for (int o = 128; o > 0; o >>= 1) { if (tid < o) red[tid] += red[tid + o]; __syncthreads(); }
  const float inv = 1.f / red[0];
  for (int i = tid; i < ncols; i += 256) r[i] *= inv;
}

__launch_bounds__(256)
__global__ void cls_pos_kernel(const float* __restrict__ v_class, float* __restrict__ x) {
  const int b = blockIdx.x;
  const int n = blockIdx.y * 256 + threadIdx.x;
  if (n >= 768) return;
  // s = 0: ang = 0 -> even dims sin(0)=0, odd dims cos(0)=1
  x[(long)b * 197 * 768 + n] = v_class[n] + ((n & 1) ? 1.f : 0.f);
}

__launch_bounds__(256)
__global__ void kan_sines_kernel(const float* __restrict__ x, float* __restrict__ sines) {
  const int b = blockIdx.x;
  for (int idx = threadIdx.x; idx < 4 * 768; idx += 256) {
    const int gi = idx / 768, i = idx % 768;
    const float xv = x[(long)b * 197 * 768 + i];
    sines[((long)b * 4 + gi) * 768 + i] =
        __sinf(xv * (float)(gi + 1) + (float)i * (3.14159265358979323846f / 768.f));
  }
}

// ---------------------------------------------------------------------------
extern "C" void kernel_launch(void* const* d_in, const int* in_sizes, int n_in,
                              void* d_out, int out_size, void* d_ws, size_t ws_size,
                              hipStream_t stream) {
  (void)in_sizes; (void)n_in; (void)out_size; (void)ws_size;

  const float* images    = (const float*)d_in[0];
  const float* v_class   = (const float*)d_in[1];
  const float* W_map     = (const float*)d_in[2];
  const float* b_map     = (const float*)d_in[3];
  const float* ln1_scale = (const float*)d_in[4];
  const float* ln1_bias  = (const float*)d_in[5];
  const float* Wqkv      = (const float*)d_in[6];
  const float* bqkv      = (const float*)d_in[7];
  const float* Wo        = (const float*)d_in[8];
  const float* bo        = (const float*)d_in[9];
  const float* ln2_scale = (const float*)d_in[10];
  const float* ln2_bias  = (const float*)d_in[11];
  const float* Wmlp1     = (const float*)d_in[12];
  const float* bmlp1     = (const float*)d_in[13];
  const float* Wmlp2     = (const float*)d_in[14];
  const float* bmlp2     = (const float*)d_in[15];
  const float* kan_amp   = (const float*)d_in[16];
  float* out = (float*)d_out;

  const int  Bsz = 64, D = 768, H = 12, NB = 12, FF = 3072, SEQ = 197;
  const long ROWS  = (long)Bsz * SEQ;       // 12608
  const long PROWS = (long)Bsz * 196;       // 12544

  // workspace (fp32)
  float* x     = (float*)d_ws;              // ROWS * D
  float* h     = x + ROWS * D;              // ROWS * D
  float* qkv   = h + ROWS * D;              // ROWS * 3D
  float* y     = qkv + ROWS * 3 * D;        // ROWS * D
  float* ff    = y + ROWS * D;              // ROWS * FF   (reused for scores)
  float* S     = ff;                        // 768 * 197 * 197 <= ROWS*FF
  float* sines = ff + ROWS * FF;            // 64 * 4 * 768

  auto G0 = []() { GemmArgs g{}; g.scale = 1.f; g.Hdiv = 1; return g; };
  const int MX = (int)((ROWS + BMT - 1) / BMT);   // 99

  // ---- patch embed: x[b,1+p,:] = patches @ W_map + b_map + pos
  {
    GemmArgs g = G0();
    g.A = images;
    g.B = W_map; g.sBk = D; g.sBn = 1;
    g.bias = b_map; g.C = x;
    g.M = (int)PROWS; g.N = D; g.K = D;
    gemm_wmma<3><<<dim3(98, 6, 1), 256, 0, stream>>>(g);
  }
  cls_pos_kernel<<<dim3(Bsz, 3), 256, 0, stream>>>(v_class, x);

  for (int nb = 0; nb < NB; nb++) {
    // LN1: x -> h
    ln_kernel<<<(unsigned)ROWS, 256, 0, stream>>>(x, ln1_scale + (long)nb * D,
                                                  ln1_bias + (long)nb * D, h);
    // QKV: h @ Wqkv + bqkv -> qkv
    {
      GemmArgs g = G0();
      g.A = h; g.sAm = D;
      g.B = Wqkv + (long)nb * D * 3 * D; g.sBk = 3 * D; g.sBn = 1;
      g.bias = bqkv + (long)nb * 3 * D;
      g.C = qkv; g.sCm = 3 * D;
      g.M = (int)ROWS; g.N = 3 * D; g.K = D;
      gemm_wmma<0><<<dim3(MX, 18, 1), 256, 0, stream>>>(g);
    }
    // scores: per (b,h) S = Q @ K^T * 1/8
    {
      GemmArgs g = G0();
      g.Hdiv = H;
      g.A = qkv; g.sAm = 3 * D; g.zAb = (long)SEQ * 3 * D; g.zAh = 64;
      g.B = qkv + D; g.sBk = 1; g.sBn = 3 * D; g.zBb = (long)SEQ * 3 * D; g.zBh = 64;
      g.C = S; g.sCm = SEQ; g.zCb = (long)H * SEQ * SEQ; g.zCh = (long)SEQ * SEQ;
      g.M = SEQ; g.N = SEQ; g.K = 64; g.scale = 0.125f;
      gemm_wmma<0><<<dim3(2, 2, Bsz * H), 256, 0, stream>>>(g);
    }
    softmax_rows<<<(unsigned)(Bsz * H * SEQ), 256, 0, stream>>>(S, SEQ);
    // y = P @ V, scattered to y[b,s, h*64 + d]
    {
      GemmArgs g = G0();
      g.Hdiv = H;
      g.A = S; g.sAm = SEQ; g.zAb = (long)H * SEQ * SEQ; g.zAh = (long)SEQ * SEQ;
      g.B = qkv + 2 * D; g.sBk = 3 * D; g.sBn = 1; g.zBb = (long)SEQ * 3 * D; g.zBh = 64;
      g.C = y; g.sCm = D; g.zCb = (long)SEQ * D; g.zCh = 64;
      g.M = SEQ; g.N = 64; g.K = SEQ;
      gemm_wmma<0><<<dim3(2, 1, Bsz * H), 256, 0, stream>>>(g);
    }
    // proj + residual: x += y @ Wo + bo
    {
      GemmArgs g = G0();
      g.A = y; g.sAm = D;
      g.B = Wo + (long)nb * D * D; g.sBk = D; g.sBn = 1;
      g.bias = bo + (long)nb * D;
      g.resid = x; g.sRm = D;
      g.C = x; g.sCm = D;
      g.M = (int)ROWS; g.N = D; g.K = D;
      gemm_wmma<2><<<dim3(MX, 6, 1), 256, 0, stream>>>(g);
    }
    // LN2: x -> h
    ln_kernel<<<(unsigned)ROWS, 256, 0, stream>>>(x, ln2_scale + (long)nb * D,
                                                  ln2_bias + (long)nb * D, h);
    // MLP1: ff = gelu(h @ Wmlp1 + bmlp1)
    {
      GemmArgs g = G0();
      g.A = h; g.sAm = D;
      g.B = Wmlp1 + (long)nb * D * FF; g.sBk = FF; g.sBn = 1;
      g.bias = bmlp1 + (long)nb * FF;
      g.C = ff; g.sCm = FF;
      g.M = (int)ROWS; g.N = FF; g.K = D;
      gemm_wmma<1><<<dim3(MX, 24, 1), 256, 0, stream>>>(g);
    }
    // MLP2 + residual: x += ff @ Wmlp2 + bmlp2
    {
      GemmArgs g = G0();
      g.A = ff; g.sAm = FF;
      g.B = Wmlp2 + (long)nb * FF * D; g.sBk = D; g.sBn = 1;
      g.bias = bmlp2 + (long)nb * D;
      g.resid = x; g.sRm = D;
      g.C = x; g.sCm = D;
      g.M = (int)ROWS; g.N = D; g.K = FF;
      gemm_wmma<2><<<dim3(MX, 6, 1), 256, 0, stream>>>(g);
    }
  }

  // ---- KAN head
  kan_sines_kernel<<<Bsz, 256, 0, stream>>>(x, sines);
  {
    GemmArgs g = G0();
    g.A = sines; g.sAm = 4 * D;
    g.B = kan_amp; g.sBk = 1; g.sBn = 4 * D;   // Bt[n][k] = kan_amp[n, k]
    g.C = out; g.sCm = 1000;
    g.M = Bsz; g.N = 1000; g.K = 4 * D;
    g.scale = 1.0f / sqrtf((float)(4 * D));
    gemm_wmma<0><<<dim3(1, 8, 1), 256, 0, stream>>>(g);
  }
  softmax_rows<<<Bsz, 256, 0, stream>>>(out, 1000);
}
